// WaveFunctionElectronUpDown_44513041056441
// MI455X (gfx1250) — matrix operand order repaired and checked by reading, not compiled, not run on hardware
//
#include <hip/hip_runtime.h>

// ---------------- problem constants ----------------
#define BB   16
#define SS   256
#define EE   1024
#define HHID 1024
#define RR   8
#define LLAY 4
#define HN_  16
#define HD_  64
#define ROWS (BB*SS)        // 4096 token rows
#define E3   (3*EE)

// ---------------- WMMA / TDM types ----------------
typedef __attribute__((ext_vector_type(16))) __bf16 v16bf;
typedef __attribute__((ext_vector_type(2)))  __bf16 v2bf;
typedef __attribute__((ext_vector_type(8)))  float  v8f;
typedef unsigned int u32x4 __attribute__((ext_vector_type(4)));
typedef int          i32x4 __attribute__((ext_vector_type(4)));
typedef int          i32x8 __attribute__((ext_vector_type(8)));

#if defined(__has_builtin)
#if __has_builtin(__builtin_amdgcn_tensor_load_to_lds) && \
    __has_builtin(__builtin_amdgcn_s_wait_tensorcnt)
#define HAVE_TDM 1
#endif
#if __has_builtin(__builtin_amdgcn_cvt_pk_bf16_f32)
#define HAVE_PKCVT 1
#endif
#endif

__device__ __forceinline__ unsigned short f2bf(float f) {
    unsigned int u = __float_as_uint(f);
    unsigned int r = (u + 0x7FFFu + ((u >> 16) & 1u)) >> 16;   // RNE
    return (unsigned short)r;
}
// pack two f32 into packed bf16 dword: 32-bit RNE rounds + one v_perm_b32
__device__ __forceinline__ unsigned int pk_bf16(float x, float y) {
#ifdef HAVE_PKCVT
    union { v2bf v; unsigned int u; } cv;
    cv.v = __builtin_amdgcn_cvt_pk_bf16_f32(x, y);
    return cv.u;
#else
    unsigned int ux = __float_as_uint(x);
    unsigned int uy = __float_as_uint(y);
    ux += 0x7FFFu + ((ux >> 16) & 1u);
    uy += 0x7FFFu + ((uy >> 16) & 1u);
    // result = { uy[31:16], ux[31:16] }  (bytes: ux.b2, ux.b3, uy.b2, uy.b3)
    return __builtin_amdgcn_perm(uy, ux, 0x07060302u);
#endif
}
__device__ __forceinline__ float gelu_f(float x) {
    return 0.5f * x * (1.0f + erff(x * 0.70710678118654752f));
}

// ---------------- generic bf16 WMMA GEMM, double-buffered ----------------
// C[M,N] = epilogue( A[M,K](f32->bf16) x Bw[K,N](bf16) )
// flags: 1=GELU  2=ACCUM(C+=)  4=+Res  8=*rowscale[m*rs_stride]
#define BM 128
#define BN 128
#define BK 32
#define KSTA 40     // A LDS k-stride (u16): 32 + 8 pad
#define NSTB 136    // B LDS n-stride (u16): 128 + 8 pad (= 272 B, matches TDM pad)

__global__ __launch_bounds__(256)
void gemm_bf16_kernel(const float* __restrict__ A,
                      const unsigned short* __restrict__ Bw,
                      const float* __restrict__ bias,
                      float* __restrict__ C,
                      const float* __restrict__ Res,
                      const float* __restrict__ rscale, int rs_stride,
                      int M, int N, int Kd, int flags)
{
    __shared__ unsigned short As[2][BM * KSTA];   // [row][k]
    __shared__ unsigned short Bs[2][BK * NSTB];   // row-major [k][n], padded rows

    const int tid  = threadIdx.x;
    const int bm0  = blockIdx.x * BM;
    const int bn0  = blockIdx.y * BN;
    const int wid  = tid >> 5, lane = tid & 31;
    const int wm   = wid & 3;      // 4 M quadrants (32 rows each)
    const int wn   = wid >> 2;     // 2 N halves   (64 cols each)
    const int laneM = lane & 15, laneH = lane >> 4;
    const int nTiles = Kd / BK;    // Kd is always a multiple of 32 here

    union FA { unsigned int u[8]; v16bf v; };
    union FC { v8f v; float f[8]; };
    FC acc[2][4];
#pragma unroll
    for (int a = 0; a < 2; ++a)
#pragma unroll
        for (int b = 0; b < 4; ++b)
#pragma unroll
            for (int r = 0; r < 8; ++r) acc[a][b].f[r] = 0.0f;

    // ---- A staging: global f32 -> regs -> bf16 LDS ----
    float4 areg[4];
    auto loadA = [&](int tile) {
#pragma unroll
        for (int it = 0; it < 4; ++it) {
            int slot = tid + it * 256;          // 1024 slots of 4 elems
            int row  = slot >> 3;
            int k0   = (slot & 7) << 2;
            const float* ap = A + (size_t)(bm0 + row) * Kd + tile * BK + k0;
            areg[it] = *(const float4*)ap;
            if ((tile + 1) * BK < Kd) __builtin_prefetch(ap + BK, 0, 0);
        }
    };
    auto storeA = [&](int buf) {
#pragma unroll
        for (int it = 0; it < 4; ++it) {
            int slot = tid + it * 256;
            int row  = slot >> 3;
            int k0   = (slot & 7) << 2;
            unsigned short* dst = &As[buf][row * KSTA + k0];
            *(unsigned int*)(dst)     = pk_bf16(areg[it].x, areg[it].y);
            *(unsigned int*)(dst + 2) = pk_bf16(areg[it].z, areg[it].w);
        }
    };

    // ---- B staging via Tensor Data Mover (wave 0 issues the DMA) ----
#ifdef HAVE_TDM
    i32x8 g1d;   // loop-invariant descriptor group 1
    {
        unsigned td0 = (unsigned)(N - bn0);     // cols remaining (OOB -> 0)
        unsigned td1 = (unsigned)Kd;            // rows never OOB (K % 32 == 0)
        g1d[0] = (int)((1u << 16) |             // data_size = 2B
                       (1u << 20) |             // pad_enable
                       (5u << 22) |             // pad_interval = 64 DWORDs
                       (3u << 25));             // pad_amount   = 4 DWORDs
        g1d[1] = (int)((td0 & 0xFFFFu) << 16);  // tensor_dim0[15:0]
        g1d[2] = (int)((td0 >> 16) | ((td1 & 0xFFFFu) << 16));
        g1d[3] = (int)((td1 >> 16) | (128u << 16));  // tile_dim0 = 128
        g1d[4] = (int)32u;                      // tile_dim1 = 32, tile_dim2 = 0
        g1d[5] = (int)N;                        // tensor_dim0_stride
        g1d[6] = 0;
        g1d[7] = 0;
    }
#endif
    auto issueB = [&](int buf, int tile) {
#ifdef HAVE_TDM
        if (tid < 32) {
            unsigned long long ga =
                (unsigned long long)(const void*)(Bw + (size_t)tile * BK * N + bn0);
            unsigned ldsa = (unsigned)(unsigned long long)(const void*)&Bs[buf][0];
            u32x4 g0;
            g0[0] = 1u;                                // count=1, user descriptor
            g0[1] = ldsa;                              // lds_addr (bytes)
            g0[2] = (unsigned)ga;                      // global_addr[31:0]
            g0[3] = (unsigned)(ga >> 32) | (2u << 30); // addr[56:32] | type=2
            i32x4 gz = {0, 0, 0, 0};
#if __clang_major__ >= 23
            i32x8 gz8 = {0, 0, 0, 0, 0, 0, 0, 0};
            __builtin_amdgcn_tensor_load_to_lds(g0, g1d, gz, gz, gz8, 0);
#else
            __builtin_amdgcn_tensor_load_to_lds(g0, g1d, gz, gz, 0);
#endif
        }
#else
        {
            int krow = tid >> 3;
            int c0   = (tid & 7) << 4;                 // 16 u16 per thread
#pragma unroll
            for (int i = 0; i < 16; ++i) {
                int n = bn0 + c0 + i;
                Bs[buf][krow * NSTB + c0 + i] =
                    (n < N) ? Bw[(size_t)(tile * BK + krow) * N + n]
                            : (unsigned short)0;
            }
        }
#endif
    };
    auto waitB = [&]() {
#ifdef HAVE_TDM
        if (tid < 32) __builtin_amdgcn_s_wait_tensorcnt(0);
#endif
    };

    // ---- pipelined main loop: stage t+1 while computing t ----
    loadA(0);
    issueB(0, 0);
    storeA(0);
    waitB();
    __syncthreads();

    for (int t = 0; t < nTiles; ++t) {
        const int cur = t & 1, nxt = cur ^ 1;
        const bool hasNext = (t + 1) < nTiles;
        if (hasNext) {
            loadA(t + 1);        // global loads in flight during compute
            issueB(nxt, t + 1);  // async DMA in flight during compute
        }

        // ---- fragments + 8 WMMAs from buffer `cur` ----
        FA fa[2], fb[4];
#pragma unroll
        for (int a = 0; a < 2; ++a) {   // A: lane=M, k halves split at lane 16
            const unsigned short* p =
                &As[cur][(wm * 32 + a * 16 + laneM) * KSTA + laneH * 8];
#pragma unroll
            for (int i = 0; i < 4; ++i) {
                fa[a].u[i]     = *(const unsigned int*)(p + 2 * i);
                fa[a].u[4 + i] = *(const unsigned int*)(p + 16 + 2 * i);
            }
        }
#pragma unroll
        for (int b = 0; b < 4; ++b) {   // B: lane = N column, K halves split at lane 16 (same k order as A)
            const unsigned short* p = &Bs[cur][wn * 64 + b * 16 + laneM];
#pragma unroll
            for (int i = 0; i < 4; ++i) {
                int k0 = laneH * 8 + 2 * i;
                fb[b].u[i]     = (unsigned int)p[k0 * NSTB] | ((unsigned int)p[(k0 + 1) * NSTB] << 16);
                fb[b].u[4 + i] = (unsigned int)p[(k0 + 16) * NSTB] | ((unsigned int)p[(k0 + 17) * NSTB] << 16);
            }
        }
#pragma unroll
        for (int a = 0; a < 2; ++a)
#pragma unroll
            for (int b = 0; b < 4; ++b)
                acc[a][b].v = __builtin_amdgcn_wmma_f32_16x16x32_bf16(
                    false, fa[a].v, false, fb[b].v, (short)0, acc[a][b].v, false, false);

        if (hasNext) storeA(nxt);   // convert + LDS store after WMMAs issued
        waitB();                    // wave 0: DMA for t+1 complete
        __syncthreads();            // everyone: buffers for t+1 visible
    }

    // ---- epilogue (C/D layout: M = r + 8*laneH, N = laneM) ----
#pragma unroll
    for (int b = 0; b < 4; ++b) {
        int ncol = bn0 + wn * 64 + b * 16 + laneM;
        if (ncol >= N) continue;
        float bv = bias ? bias[ncol] : 0.0f;
#pragma unroll
        for (int a = 0; a < 2; ++a) {
            int mbase = bm0 + wm * 32 + a * 16 + laneH * 8;
#pragma unroll
            for (int r = 0; r < 8; ++r) {
                int m = mbase + r;
                float v = acc[a][b].f[r] + bv;
                if (flags & 1) v = gelu_f(v);
                if (flags & 8) v *= rscale[(size_t)m * rs_stride];
                size_t idx = (size_t)m * N + ncol;
                if (flags & 2) v += C[idx];
                if (flags & 4) v += Res[idx];
                C[idx] = v;
            }
        }
    }
}

// ---------------- f32 -> bf16 weight cast (2 elems -> 1 dword) ----------------
__global__ void cast_bf16_kernel(const float* __restrict__ src,
                                 unsigned int* __restrict__ dst, long npair)
{
    long i = (long)blockIdx.x * blockDim.x + threadIdx.x;
    long stride = (long)gridDim.x * blockDim.x;
    for (; i < npair; i += stride) {
        float2 f = *(const float2*)(src + 2 * i);
        dst[i] = pk_bf16(f.x, f.y);
    }
}

// ---------------- embedding gather ----------------
__global__ void embed_kernel(const int* __restrict__ x,
                             const float* __restrict__ emb,
                             float* __restrict__ xc)
{
    int row = blockIdx.x;            // b*S + s
    int s   = row % SS;
    int cls = x[row];
    const float* src = emb + ((size_t)s * 4 + cls) * EE;
    float* dst = xc + (size_t)row * EE;
    for (int e = threadIdx.x; e < EE; e += 256) dst[e] = src[e];
}

// ---------------- causal attention (per b,h,q) ----------------
__global__ __launch_bounds__(64)
void attn_kernel(const float* __restrict__ qkv, float* __restrict__ out)
{
    __shared__ float qv[HD_];
    __shared__ float sc[SS];
    __shared__ float red[64];
    int blk = blockIdx.x;
    int q = blk % SS;
    int h = (blk / SS) % HN_;
    int b = blk / (SS * HN_);
    int t = threadIdx.x;
    size_t rowq = (size_t)(b * SS + q);

    qv[t] = qkv[rowq * E3 + h * HD_ + t];
    __syncthreads();

    for (int j = t; j <= q; j += 64) {
        const float* kr = qkv + (size_t)(b * SS + j) * E3 + EE + h * HD_;
        float s = 0.0f;
#pragma unroll 8
        for (int d = 0; d < HD_; ++d) s += qv[d] * kr[d];
        sc[j] = s * 0.125f;    // HD^-0.5
    }
    __syncthreads();

    float mx = -1e30f;
    for (int j = t; j <= q; j += 64) mx = fmaxf(mx, sc[j]);
    red[t] = mx;
    for (int off = 32; off > 0; off >>= 1) {
        __syncthreads();
        if (t < off) red[t] = fmaxf(red[t], red[t + off]);
    }
    __syncthreads();
    mx = red[0];
    __syncthreads();

    float sum = 0.0f;
    for (int j = t; j <= q; j += 64) { float e = __expf(sc[j] - mx); sc[j] = e; sum += e; }
    red[t] = sum;
    for (int off = 32; off > 0; off >>= 1) {
        __syncthreads();
        if (t < off) red[t] += red[t + off];
    }
    __syncthreads();
    float inv = 1.0f / red[0];
    __syncthreads();

    float o = 0.0f;           // t == output dim d
    for (int j = 0; j <= q; ++j) {
        const float* vr = qkv + (size_t)(b * SS + j) * E3 + 2 * EE + h * HD_;
        o += sc[j] * vr[t];
    }
    out[rowq * EE + h * HD_ + t] = o * inv;
}

// ---------------- layernorm ----------------
__global__ __launch_bounds__(256)
void ln_kernel(const float* __restrict__ in, const float* __restrict__ sc,
               const float* __restrict__ bt, float* __restrict__ out)
{
    __shared__ float red[256];
    int row = blockIdx.x, t = threadIdx.x;
    const float* x = in + (size_t)row * EE;
    float s = 0.0f;
    for (int e = t; e < EE; e += 256) s += x[e];
    red[t] = s;
    for (int off = 128; off > 0; off >>= 1) { __syncthreads(); if (t < off) red[t] += red[t + off]; }
    __syncthreads();
    float mean = red[0] * (1.0f / EE);
    __syncthreads();
    s = 0.0f;
    for (int e = t; e < EE; e += 256) { float d = x[e] - mean; s += d * d; }
    red[t] = s;
    for (int off = 128; off > 0; off >>= 1) { __syncthreads(); if (t < off) red[t] += red[t + off]; }
    __syncthreads();
    float inv = rsqrtf(red[0] * (1.0f / EE) + 1e-5f);
    float* o = out + (size_t)row * EE;
    for (int e = t; e < EE; e += 256) o[e] = (x[e] - mean) * inv * sc[e] + bt[e];
}

// ---------------- router: softmax over 8 centroids, top-2 gate ----------------
__global__ __launch_bounds__(64)
void route_kernel(const float* __restrict__ xc, const float* __restrict__ cent,
                  float* __restrict__ gate)
{
    __shared__ float red[RR][64];
    __shared__ float sim[RR];
    int row = blockIdx.x, t = threadIdx.x;
    const float* x = xc + (size_t)row * EE;
    float a[RR];
#pragma unroll
    for (int r = 0; r < RR; ++r) a[r] = 0.0f;
    for (int e = t; e < EE; e += 64) {
        float xv = x[e];
#pragma unroll
        for (int r = 0; r < RR; ++r) a[r] += xv * cent[r * EE + e];
    }
#pragma unroll
    for (int r = 0; r < RR; ++r) red[r][t] = a[r];
    __syncthreads();
    if (t < RR) {
        float s = 0.0f;
        for (int i = 0; i < 64; ++i) s += red[t][i];
        sim[t] = s;
    }
    __syncthreads();
    if (t == 0) {
        float mx = sim[0];
        for (int r = 1; r < RR; ++r) mx = fmaxf(mx, sim[r]);
        float p[RR], sum = 0.0f;
        for (int r = 0; r < RR; ++r) { p[r] = __expf(sim[r] - mx); sum += p[r]; }
        for (int r = 0; r < RR; ++r) p[r] /= sum;
        int i1 = 0;
        for (int r = 1; r < RR; ++r) if (p[r] > p[i1]) i1 = r;
        int i2 = (i1 == 0) ? 1 : 0;
        for (int r = 0; r < RR; ++r) if (r != i1 && p[r] > p[i2]) i2 = r;
        float g = p[i1] + p[i2];
        for (int r = 0; r < RR; ++r)
            gate[(size_t)row * RR + r] = (r == i1 || r == i2) ? p[r] / g : 0.0f;
    }
}

// ---------------- y += sum_r gate[row,r] * r_b2[r,:] ----------------
__global__ void b2add_kernel(float* __restrict__ y, const float* __restrict__ gate,
                             const float* __restrict__ rb2)
{
    int row = blockIdx.x;
    float g[RR];
#pragma unroll
    for (int r = 0; r < RR; ++r) g[r] = gate[(size_t)row * RR + r];
    for (int e = threadIdx.x; e < EE; e += 256) {
        float s = 0.0f;
#pragma unroll
        for (int r = 0; r < RR; ++r) s += g[r] * rb2[(size_t)r * EE + e];
        y[(size_t)row * EE + e] += s;
    }
}

// ---------------- host orchestration ----------------
extern "C" void kernel_launch(void* const* d_in, const int* in_sizes, int n_in,
                              void* d_out, int out_size, void* d_ws, size_t ws_size,
                              hipStream_t stream)
{
    const int*   x      = (const int*)  d_in[0];
    const float* emb    = (const float*)d_in[1];
    const float* qkv_w  = (const float*)d_in[2];
    const float* qkv_b  = (const float*)d_in[3];
    const float* out_w  = (const float*)d_in[4];
    const float* out_b  = (const float*)d_in[5];
    const float* ln1_s  = (const float*)d_in[6];
    const float* ln1_b  = (const float*)d_in[7];
    const float* sh_w1  = (const float*)d_in[8];
    const float* sh_b1  = (const float*)d_in[9];
    const float* sh_w2  = (const float*)d_in[10];
    const float* sh_b2  = (const float*)d_in[11];
    const float* cent   = (const float*)d_in[12];
    const float* r_w1   = (const float*)d_in[13];
    const float* r_b1   = (const float*)d_in[14];
    const float* r_w2   = (const float*)d_in[15];
    const float* r_b2   = (const float*)d_in[16];
    const float* ln2_s  = (const float*)d_in[17];
    const float* ln2_b  = (const float*)d_in[18];
    const float* t_w1   = (const float*)d_in[19];
    const float* t_b1   = (const float*)d_in[20];
    const float* t_w2   = (const float*)d_in[21];
    const float* t_b2   = (const float*)d_in[22];

    char* w = (char*)d_ws;
    size_t off = 0;
    auto alloc = [&](size_t bytes) -> char* {
        char* p = w + off;
        off = (off + bytes + 255) & ~(size_t)255;
        return p;
    };
    unsigned short* qkvw_b = (unsigned short*)alloc((size_t)LLAY * EE * E3   * 2);
    unsigned short* outw_b = (unsigned short*)alloc((size_t)LLAY * EE * EE   * 2);
    unsigned short* shw1_b = (unsigned short*)alloc((size_t)LLAY * EE * HHID * 2);
    unsigned short* shw2_b = (unsigned short*)alloc((size_t)LLAY * HHID * EE * 2);
    unsigned short* rw1_b  = (unsigned short*)alloc((size_t)LLAY * RR * EE * HHID * 2);
    unsigned short* rw2_b  = (unsigned short*)alloc((size_t)LLAY * RR * HHID * EE * 2);
    unsigned short* tw1_b  = (unsigned short*)alloc((size_t)EE * HHID * 2);
    unsigned short* tw2_b  = (unsigned short*)alloc((size_t)HHID * 8 * 2);
    float* xc     = (float*)alloc((size_t)ROWS * EE * 4);
    float* qkvbuf = (float*)alloc((size_t)ROWS * E3 * 4);
    float* hbuf   = (float*)alloc((size_t)ROWS * HHID * 4);
    float* ybuf   = (float*)alloc((size_t)ROWS * EE * 4);
    float* gbuf   = (float*)alloc((size_t)ROWS * RR * 4);

    auto cast = [&](const float* s, unsigned short* d, long n) {
        cast_bf16_kernel<<<1024, 256, 0, stream>>>(s, (unsigned int*)d, n / 2);
    };
    cast(qkv_w, qkvw_b, (long)LLAY * EE * E3);
    cast(out_w, outw_b, (long)LLAY * EE * EE);
    cast(sh_w1, shw1_b, (long)LLAY * EE * HHID);
    cast(sh_w2, shw2_b, (long)LLAY * HHID * EE);
    cast(r_w1,  rw1_b,  (long)LLAY * RR * EE * HHID);
    cast(r_w2,  rw2_b,  (long)LLAY * RR * HHID * EE);
    cast(t_w1,  tw1_b,  (long)EE * HHID);
    cast(t_w2,  tw2_b,  (long)HHID * 8);

    auto gemm = [&](const float* A, const unsigned short* Bw, const float* bias,
                    float* C, const float* Res, const float* rs, int rss,
                    int M, int N, int Kd, int flags) {
        dim3 g(M / BM, (N + BN - 1) / BN);
        gemm_bf16_kernel<<<g, 256, 0, stream>>>(A, Bw, bias, C, Res, rs, rss,
                                                M, N, Kd, flags);
    };

    embed_kernel<<<ROWS, 256, 0, stream>>>(x, emb, xc);

    for (int l = 0; l < LLAY; ++l) {
        // qkv projection
        gemm(xc, qkvw_b + (size_t)l * EE * E3, qkv_b + (size_t)l * E3,
             qkvbuf, nullptr, nullptr, 0, ROWS, E3, EE, 0);
        // attention
        attn_kernel<<<BB * HN_ * SS, 64, 0, stream>>>(qkvbuf, hbuf);
        // out proj + residual
        gemm(hbuf, outw_b + (size_t)l * EE * EE, out_b + (size_t)l * EE,
             ybuf, xc, nullptr, 0, ROWS, EE, EE, 4);
        ln_kernel<<<ROWS, 256, 0, stream>>>(ybuf, ln1_s + (size_t)l * EE,
                                            ln1_b + (size_t)l * EE, xc);
        // shared FFN (NSH == 1)
        gemm(xc, shw1_b + (size_t)l * EE * HHID, sh_b1 + (size_t)l * HHID,
             hbuf, nullptr, nullptr, 0, ROWS, HHID, EE, 1);
        gemm(hbuf, shw2_b + (size_t)l * HHID * EE, sh_b2 + (size_t)l * EE,
             ybuf, xc, nullptr, 0, ROWS, EE, HHID, 4);
        // routing gate
        route_kernel<<<ROWS, 64, 0, stream>>>(xc, cent + (size_t)l * RR * EE, gbuf);
        // dense experts, gate folded into hidden activations
        for (int r = 0; r < RR; ++r) {
            gemm(xc, rw1_b + ((size_t)l * RR + r) * EE * HHID,
                 r_b1 + ((size_t)l * RR + r) * HHID,
                 hbuf, nullptr, gbuf + r, RR, ROWS, HHID, EE, 1 | 8);
            gemm(hbuf, rw2_b + ((size_t)l * RR + r) * HHID * EE, nullptr,
                 ybuf, nullptr, nullptr, 0, ROWS, EE, HHID, 2);
        }
        b2add_kernel<<<ROWS, 256, 0, stream>>>(ybuf, gbuf, r_b2 + (size_t)l * RR * EE);
        ln_kernel<<<ROWS, 256, 0, stream>>>(ybuf, ln2_s + (size_t)l * EE,
                                            ln2_b + (size_t)l * EE, xc);
    }

    // head
    gemm(xc, tw1_b, t_b1, hbuf, nullptr, nullptr, 0, ROWS, HHID, EE, 1);
    gemm(hbuf, tw2_b, t_b2, (float*)d_out, nullptr, nullptr, 0, ROWS, 8, HHID, 0);
}
